// DecoderAttention_62989990363717
// MI455X (gfx1250) — matrix operand order repaired and checked by reading, not compiled, not run on hardware
//
#include <hip/hip_runtime.h>

// DecoderAttention for MI455X (gfx1250, wave32, WMMA).
// Fused flash-attention: enc_out (168 MB) is read exactly once from HBM.
// All matmuls: v_wmma_f32_16x16x32_f16 (f16 in, fp32 accumulate); softmax fp32.

#define B_ 256
#define S_ 2048
#define P_ 64
#define D_ 80
#define SC 64            // S-chunk per iteration

typedef __attribute__((ext_vector_type(16))) _Float16 v16h;
typedef __attribute__((ext_vector_type(8)))  _Float16 v8h;
typedef __attribute__((ext_vector_type(8)))  float    v8f;

#define PITCH_RM 104     // halves/row for row-major K-padded buffers (80->96 pad, 16B-aligned, not pow2)
#define PITCH_T  72      // halves/row for transposed-enc and P buffers

// Shared memory regions (bytes). rm/tp regions are reused across phases.
#define OFF_RM   0        // phase0: W (80x104) | main: E_rm (64x104)
#define OFF_T    16640    // phase0: dec (64x104) | main: E_t (80x72)
#define OFF_Q    29952    // Q (64x104)
#define OFF_P    43264    // P tiles, 4 waves x 16x72
#define SMEM_BYTES 52480

__device__ __forceinline__ v8f wmma_f16(v16h a, v16h b, v8f c) {
  // 8 args: (neg_a, A, neg_b, B, c_mod, C, reuse_a, reuse_b)
  return __builtin_amdgcn_wmma_f32_16x16x32_f16(false, a, false, b, (short)0, c, false, false);
}

// A-matrix fragment (16xK tile, row-major source).
// ISA layout: lane L holds row M=L%16; element i -> K = (i<8 ? i : i+8) + 8*(L/16).
__device__ __forceinline__ v16h load_A(const _Float16* base, int pitch, int row0, int kbase, int lane) {
  const int hlf = lane >> 4;
  const _Float16* p = base + (row0 + (lane & 15)) * pitch + kbase + hlf * 8;
  v8h lo = *(const v8h*)(p);        // K = kbase + 8*hlf + 0..7
  v8h hi = *(const v8h*)(p + 16);   // K = kbase + 16 + 8*hlf + 0..7
  v16h a;
#pragma unroll
  for (int i = 0; i < 8; ++i) { a[i] = lo[i]; a[i + 8] = hi[i]; }
  return a;
}

// B-matrix fragment (Kx16 tile) from a source where B[k][n] = src[nrow0+n][k]
// (i.e. k's contiguous in the source row). ISA layout: N=L%16; element i -> K = (i<8 ? i : i+8) + 8*(L/16).
__device__ __forceinline__ v16h load_B(const _Float16* base, int pitch, int nrow0, int kbase, int lane) {
  const int hlf = lane >> 4;
  const _Float16* p = base + (nrow0 + (lane & 15)) * pitch + kbase + hlf * 8;
  v8h lo = *(const v8h*)(p);
  v8h hi = *(const v8h*)(p + 16);
  v16h b;
#pragma unroll
  for (int i = 0; i < 8; ++i) { b[i] = lo[i]; b[i + 8] = hi[i]; }
  return b;
}

__global__ __launch_bounds__(128, 1)
void decoder_attn_gfx1250(const float* __restrict__ enc,
                          const float* __restrict__ dec,
                          const int*   __restrict__ mask,
                          const float* __restrict__ W,
                          float*       __restrict__ out) {
  __shared__ alignas(16) char smem[SMEM_BYTES];
  _Float16* rm   = (_Float16*)(smem + OFF_RM);
  _Float16* tp   = (_Float16*)(smem + OFF_T);
  _Float16* qld  = (_Float16*)(smem + OFF_Q);
  _Float16* pbuf = (_Float16*)(smem + OFF_P);

  const int b    = blockIdx.x;
  const int tid  = threadIdx.x;
  const int wave = tid >> 5;      // wave w owns query rows 16w..16w+15
  const int lane = tid & 31;
  const int hlf  = lane >> 4;
  const int n    = lane & 15;

  // ---------- Phase 0a: stage W (80x80) and dec tile (64x80) as f16, zero-pad K to 96 ----------
  for (int idx = tid; idx < D_ * D_; idx += 128) {
    int e = idx / D_, d = idx - e * D_;
    rm[e * PITCH_RM + d] = (_Float16)W[idx];
  }
  for (int idx = tid; idx < D_ * (PITCH_RM - D_); idx += 128) {
    int e = idx / (PITCH_RM - D_), j = idx - e * (PITCH_RM - D_);
    rm[e * PITCH_RM + D_ + j] = (_Float16)0.0f;
  }
  const int lrow = tid >> 1, lh2 = tid & 1;   // cooperative loader: 2 threads per row
  {
    const float4* src4 = (const float4*)(dec + ((long)b * P_ + lrow) * D_ + lh2 * 40);
#pragma unroll
    for (int j4 = 0; j4 < 10; ++j4) {
      float4 v = src4[j4];
      int d0 = lh2 * 40 + j4 * 4;
      tp[lrow * PITCH_RM + d0 + 0] = (_Float16)v.x;
      tp[lrow * PITCH_RM + d0 + 1] = (_Float16)v.y;
      tp[lrow * PITCH_RM + d0 + 2] = (_Float16)v.z;
      tp[lrow * PITCH_RM + d0 + 3] = (_Float16)v.w;
    }
    if (lh2) {
#pragma unroll
      for (int j = 0; j < PITCH_RM - D_; ++j)
        tp[lrow * PITCH_RM + D_ + j] = (_Float16)0.0f;
    }
  }
  __syncthreads();

  // ---------- Phase 0b: Q[p,e] = sum_d dec[p,d] * W[e,d]  (per-wave 16x80) ----------
  {
    v8f q[5];
#pragma unroll
    for (int nt = 0; nt < 5; ++nt)
#pragma unroll
      for (int r = 0; r < 8; ++r) q[nt][r] = 0.0f;
#pragma unroll
    for (int kt = 0; kt < 3; ++kt) {
      v16h a = load_A(tp, PITCH_RM, wave * 16, kt * 32, lane);
#pragma unroll
      for (int nt = 0; nt < 5; ++nt) {
        v16h bf = load_B(rm, PITCH_RM, nt * 16, kt * 32, lane);  // B[k][n] = W[e=n][d=k]
        q[nt] = wmma_f16(a, bf, q[nt]);
      }
    }
    // C-layout -> row-major LDS (f16), zero-pad K cols 80..103
    _Float16* qrow = qld + (wave * 16 + hlf * 8) * PITCH_RM;
#pragma unroll
    for (int r = 0; r < 8; ++r) {
#pragma unroll
      for (int nt = 0; nt < 5; ++nt)
        qrow[r * PITCH_RM + nt * 16 + n] = (_Float16)q[nt][r];
      qrow[r * PITCH_RM + D_ + n] = (_Float16)0.0f;
      if (n < 8) qrow[r * PITCH_RM + 96 + n] = (_Float16)0.0f;
    }
  }

  // ---------- Main loop: online-softmax attention over S in chunks of 64 ----------
  float mrun[8], lrun[8];
  v8f ctx[5];
#pragma unroll
  for (int r = 0; r < 8; ++r) { mrun[r] = -3.0e38f; lrun[r] = 0.0f; }
#pragma unroll
  for (int nt = 0; nt < 5; ++nt)
#pragma unroll
    for (int r = 0; r < 8; ++r) ctx[nt][r] = 0.0f;

  const int* mrow_base = mask + (long)b * S_;

  for (int c = 0; c < S_ / SC; ++c) {
    __syncthreads();  // previous chunk fully consumed before overwrite
    // Stage enc chunk: rm = row-major (keys x d, K-padded), tp = transposed (d x keys)
    {
      const float4* src4 = (const float4*)(enc + ((long)b * S_ + c * SC + lrow) * D_ + lh2 * 40);
#pragma unroll
      for (int j4 = 0; j4 < 10; ++j4) {
        float4 v = src4[j4];
        int d0 = lh2 * 40 + j4 * 4;
        rm[lrow * PITCH_RM + d0 + 0] = (_Float16)v.x;
        rm[lrow * PITCH_RM + d0 + 1] = (_Float16)v.y;
        rm[lrow * PITCH_RM + d0 + 2] = (_Float16)v.z;
        rm[lrow * PITCH_RM + d0 + 3] = (_Float16)v.w;
        tp[(d0 + 0) * PITCH_T + lrow] = (_Float16)v.x;
        tp[(d0 + 1) * PITCH_T + lrow] = (_Float16)v.y;
        tp[(d0 + 2) * PITCH_T + lrow] = (_Float16)v.z;
        tp[(d0 + 3) * PITCH_T + lrow] = (_Float16)v.w;
      }
      if (lh2) {
#pragma unroll
        for (int j = 0; j < PITCH_RM - D_; ++j)
          rm[lrow * PITCH_RM + D_ + j] = (_Float16)0.0f;
      }
      if (c + 1 < S_ / SC)   // pull next chunk toward L2/WGP$ (global_prefetch_b8)
        __builtin_prefetch(enc + ((long)b * S_ + (c + 1) * SC + lrow) * D_ + lh2 * 40, 0, 3);
    }
    __syncthreads();

    // Scores: 16x64 strip per wave, K = 96 (padded)
    v8f sc[4];
#pragma unroll
    for (int st = 0; st < 4; ++st)
#pragma unroll
      for (int r = 0; r < 8; ++r) sc[st][r] = 0.0f;
#pragma unroll
    for (int kt = 0; kt < 3; ++kt) {
      v16h a = load_A(qld, PITCH_RM, wave * 16, kt * 32, lane);
#pragma unroll
      for (int st = 0; st < 4; ++st) {
        v16h bf = load_B(rm, PITCH_RM, st * 16, kt * 32, lane);  // B[k][n] = E[s=n][d=k]
        sc[st] = wmma_f16(a, bf, sc[st]);
      }
    }

    // Mask: column s masked -> huge negative (finite, so exp() underflows to 0, no NaN)
#pragma unroll
    for (int st = 0; st < 4; ++st) {
      int mv = mrow_base[c * SC + st * 16 + n];
      if (mv) {
#pragma unroll
        for (int r = 0; r < 8; ++r) sc[st][r] = -3.4e38f;
      }
    }

    // Online softmax per row (row M = r + 8*hlf lives in one 16-lane half-group)
    _Float16* prow = pbuf + (wave * 16 + hlf * 8) * PITCH_T;
#pragma unroll
    for (int r = 0; r < 8; ++r) {
      float cmax = fmaxf(fmaxf(sc[0][r], sc[1][r]), fmaxf(sc[2][r], sc[3][r]));
#pragma unroll
      for (int off = 1; off < 16; off <<= 1)
        cmax = fmaxf(cmax, __shfl_xor(cmax, off, 32));
      float mnew  = fmaxf(mrun[r], cmax);
      float scale = __expf(mrun[r] - mnew);
      mrun[r] = mnew;
      float rsum = 0.0f;
#pragma unroll
      for (int st = 0; st < 4; ++st) {
        float p = __expf(sc[st][r] - mnew);
        rsum += p;
        prow[r * PITCH_T + st * 16 + n] = (_Float16)p;   // C-layout -> A-layout via LDS
      }
#pragma unroll
      for (int off = 1; off < 16; off <<= 1)
        rsum += __shfl_xor(rsum, off, 32);
      lrun[r] = lrun[r] * scale + rsum;
#pragma unroll
      for (int nt = 0; nt < 5; ++nt) ctx[nt][r] *= scale;
    }

    // ctx += P(16x64) @ E(64x80): B[k][n] = E[s=k][d=n] = tp[n][k] (contiguous)
    const _Float16* pa = pbuf + wave * 16 * PITCH_T;
#pragma unroll
    for (int kt = 0; kt < 2; ++kt) {
      v16h a = load_A(pa, PITCH_T, 0, kt * 32, lane);
#pragma unroll
      for (int nt = 0; nt < 5; ++nt) {
        v16h bf = load_B(tp, PITCH_T, nt * 16, kt * 32, lane);
        ctx[nt] = wmma_f16(a, bf, ctx[nt]);
      }
    }
  }

  // ---------- Finalize: divide by softmax sum, store fp32 ----------
  float* orow = out + ((long)b * P_ + wave * 16 + hlf * 8) * D_;
#pragma unroll
  for (int r = 0; r < 8; ++r) {
    float inv = 1.0f / lrun[r];
#pragma unroll
    for (int nt = 0; nt < 5; ++nt)
      orow[r * D_ + nt * 16 + n] = ctx[nt][r] * inv;
  }
}

extern "C" void kernel_launch(void* const* d_in, const int* in_sizes, int n_in,
                              void* d_out, int out_size, void* d_ws, size_t ws_size,
                              hipStream_t stream) {
  const float* enc = (const float*)d_in[0];   // (B,S,D) f32
  const float* dec = (const float*)d_in[1];   // (B,P,D) f32
  const int*   msk = (const int*)d_in[2];     // (B,S)   nonzero = masked
  const float* W   = (const float*)d_in[3];   // (D,D)   f32
  float* out = (float*)d_out;                 // (B,P,D) f32
  (void)in_sizes; (void)n_in; (void)out_size; (void)d_ws; (void)ws_size;

  hipLaunchKernelGGL(decoder_attn_gfx1250, dim3(B_), dim3(128), 0, stream,
                     enc, dec, msk, W, out);
}
